// GEChebNet_35562329211077
// MI455X (gfx1250) — matrix operand order repaired and checked by reading, not compiled, not run on hardware
//
#include <hip/hip_runtime.h>

typedef __attribute__((ext_vector_type(16))) __bf16 v16bf;
typedef __attribute__((ext_vector_type(8)))  __bf16 v8bf;
typedef __attribute__((ext_vector_type(8)))  float  v8f;

#define NV   16384            // graph nodes
#define NB   8                // batch
#define NM   (NV * NB)        // GEMM rows M = 131072
#define KCHEB 6

#define ASTRIDE 48            // LDS row stride (bf16) for A tile: 96B, kills 4-way conflicts
#define BSTRIDE 48            // LDS row stride (bf16) for B tile

// ---------------- input transform: x[B,3,V] -> T0[V][B][Cpad=32] (zero padded) --------
__global__ void transform_input_kernel(const float* __restrict__ x, float* __restrict__ out,
                                       int Cpad) {
  size_t idx = (size_t)blockIdx.x * blockDim.x + threadIdx.x;
  size_t total = (size_t)NM * Cpad;
  if (idx >= total) return;
  int c = (int)(idx % Cpad);
  int b = (int)((idx / Cpad) % NB);
  int v = (int)(idx / ((size_t)Cpad * NB));
  out[idx] = (c < 3) ? x[((size_t)b * 3 + c) * NV + v] : 0.0f;
}

// ------- weights: fp32 [K,Cin,Cout] -> bf16 TRANSPOSED [K][Npad][Cpad] zero padded ----
// WpT[kc][n][c] = W[kc][c][n]; contiguous in c (= GEMM reduction dim K)
__global__ void pad_weights_kernel(const float* __restrict__ W, __bf16* __restrict__ WpT,
                                   int Cin, int Cout, int Cpad, int Npad) {
  size_t idx = (size_t)blockIdx.x * blockDim.x + threadIdx.x;
  size_t total = (size_t)KCHEB * Cpad * Npad;
  if (idx >= total) return;
  int c  = (int)(idx % Cpad);
  int n  = (int)((idx / Cpad) % Npad);
  int kc = (int)(idx / ((size_t)Cpad * Npad));
  float w = (c < Cin && n < Cout) ? W[((size_t)kc * Cin + c) * Cout + n] : 0.0f;
  WpT[idx] = (__bf16)w;
}

// ---------------- SpMM: Y[row] += val * X[col]   (X,Y : [V][ldc] fp32, L2-resident) ---
__global__ void spmm_kernel(const int* __restrict__ rows, const int* __restrict__ cols,
                            const float* __restrict__ vals, const float* __restrict__ X,
                            float* __restrict__ Y, int ldc, int E) {
  long long idx = (long long)blockIdx.x * blockDim.x + threadIdx.x;
  int per = ldc >> 2;                       // float4 chunks per edge
  long long total = (long long)E * per;
  if (idx >= total) return;
  int e  = (int)(idx / per);
  int c4 = (int)(idx % per) << 2;
  int r = rows[e], cI = cols[e];
  float v = vals[e];
  float4 xv = *(const float4*)(X + (size_t)cI * ldc + c4);
  float* yp = Y + (size_t)r * ldc + c4;
  atomicAdd(yp + 0, v * xv.x);
  atomicAdd(yp + 1, v * xv.y);
  atomicAdd(yp + 2, v * xv.z);
  atomicAdd(yp + 3, v * xv.w);
}

// ---------------- Chebyshev recurrence combine: Z = 2*Z - T0 -------------------------
__global__ void cheb_combine_kernel(float* __restrict__ Z, const float* __restrict__ T0,
                                    size_t n4) {
  size_t idx = (size_t)blockIdx.x * blockDim.x + threadIdx.x;
  if (idx >= n4) return;
  float4 z = ((const float4*)Z)[idx];
  float4 t = ((const float4*)T0)[idx];
  z.x = 2.0f * z.x - t.x; z.y = 2.0f * z.y - t.y;
  z.z = 2.0f * z.z - t.z; z.w = 2.0f * z.w - t.w;
  ((float4*)Z)[idx] = z;
}

// ---------------- bf16 WMMA GEMM: Cacc[M,N] += cvt_bf16(A[M,K]) @ W[K,N] --------------
// Block tile 128(M) x 64(N), 256 threads = 8 wave32.
// Wave w owns rows w*16..w*16+15 and ALL 64 cols: 4 accumulators -> 4 WMMA per K-step,
// A fragment loaded once per K-step and reused 4x. Weights (WpT[n][k]) staged whole
// into LDS before the K-loop (<=16KB), so the loop has a single barrier pair for A.
// Optional fused epilogue: += bias, ReLU (used on the last Chebyshev-order GEMM).
__global__ __launch_bounds__(256) void cheb_gemm_wmma(
    const float* __restrict__ A, const __bf16* __restrict__ WbT,
    float* __restrict__ Cacc, int K, int N,
    const float* __restrict__ bias, int Nreal) {
  __shared__ __bf16 As[128][ASTRIDE];       // A tile (only cols 0..31 used)
  __shared__ __bf16 Bs[4][64][BSTRIDE];     // all K pages of B: [kpage][n][k]
  const int tid  = threadIdx.x;
  const int wave = tid >> 5, lane = tid & 31;
  const int hi   = lane >> 4, l16 = lane & 15;
  const int m0 = blockIdx.x * 128;
  const int n0 = blockIdx.y * 64;
  const int wrow = wave * 16;
  const int kpages = K >> 5;

  // ---- stage ALL of B once: per page 64 n x 32 k; thread -> (n = tid/4, k = tid%4*8)
  {
    const int bn = tid >> 2;
    const int bk = (tid & 3) * 8;
    for (int pg = 0; pg < kpages; ++pg) {
      const __bf16* src = WbT + (size_t)(n0 + bn) * K + pg * 32 + bk;
      *(v8bf*)&Bs[pg][bn][bk] = *(const v8bf*)src;
    }
  }

  // ---- load accumulator fragments (ISA C layout: VGPR r -> M = 8*hi + r, N = lane%16)
  v8f cf[4];
#pragma unroll
  for (int nt = 0; nt < 4; ++nt) {
    const int col = n0 + nt * 16 + l16;
#pragma unroll
    for (int r = 0; r < 8; ++r)
      cf[nt][r] = Cacc[(size_t)(m0 + wrow + hi * 8 + r) * N + col];
  }

  const int lr = tid >> 1;            // 0..127 : A staging row
  const int lk = (tid & 1) * 16;      // 0 / 16 : A staging k-offset

  for (int pg = 0; pg < kpages; ++pg) {
    __syncthreads();                  // protects As reuse (and B on first iter)
    // stage A (fp32 -> bf16): 128 x 32, 16 elements per thread
    const float* ap = A + (size_t)(m0 + lr) * K + pg * 32 + lk;
    if (pg + 1 < kpages)
      __builtin_prefetch(ap + 32, 0, 1);   // global_prefetch_b8: next A page
    float4 f0 = *(const float4*)(ap + 0);
    float4 f1 = *(const float4*)(ap + 4);
    float4 f2 = *(const float4*)(ap + 8);
    float4 f3 = *(const float4*)(ap + 12);
    v8bf plo, phi;
    plo[0] = (__bf16)f0.x; plo[1] = (__bf16)f0.y; plo[2] = (__bf16)f0.z; plo[3] = (__bf16)f0.w;
    plo[4] = (__bf16)f1.x; plo[5] = (__bf16)f1.y; plo[6] = (__bf16)f1.z; plo[7] = (__bf16)f1.w;
    phi[0] = (__bf16)f2.x; phi[1] = (__bf16)f2.y; phi[2] = (__bf16)f2.z; phi[3] = (__bf16)f2.w;
    phi[4] = (__bf16)f3.x; phi[5] = (__bf16)f3.y; phi[6] = (__bf16)f3.z; phi[7] = (__bf16)f3.w;
    *(v8bf*)&As[lr][lk + 0] = plo;
    *(v8bf*)&As[lr][lk + 8] = phi;
    __syncthreads();

    // A fragment (16-bit A 16x32): elems 0..7 -> K=hi*16+j ; 8..15 -> K=hi*16+8+j
    v16bf a;
    v8bf alo = *(const v8bf*)&As[wrow + l16][hi * 16];
    v8bf ahi = *(const v8bf*)&As[wrow + l16][hi * 16 + 8];
#pragma unroll
    for (int j = 0; j < 8; ++j) { a[j] = alo[j]; a[8 + j] = ahi[j]; }

#pragma unroll
    for (int nt = 0; nt < 4; ++nt) {
      // B fragment (B 32x16, same k order as A): elem e -> K = hi*16 + e ; N = lane%16
      v16bf b = *(const v16bf*)&Bs[pg][nt * 16 + l16][hi * 16];
      cf[nt] = __builtin_amdgcn_wmma_f32_16x16x32_bf16(false, a, false, b, (short)0,
                                                       cf[nt], false, false);
    }
  }

  // ---- store (optionally fused bias + ReLU on the final Chebyshev-order GEMM)
#pragma unroll
  for (int nt = 0; nt < 4; ++nt) {
    const int col = n0 + nt * 16 + l16;
    float badd = 0.0f;
    if (bias) badd = (col < Nreal) ? bias[col] : 0.0f;
#pragma unroll
    for (int r = 0; r < 8; ++r) {
      float v = cf[nt][r];
      if (bias) { v += badd; v = v > 0.0f ? v : 0.0f; }
      Cacc[(size_t)(m0 + wrow + hi * 8 + r) * N + col] = v;
    }
  }
}

// ---------------- BN stats: per-channel sum / sumsq over all NM rows ------------------
__global__ void bn_stats_kernel(const float* __restrict__ H, float* __restrict__ sums,
                                int F) {
  int f = threadIdx.x;                       // blockDim.x == F (64 or 128)
  size_t r0 = (size_t)blockIdx.x * 256;
  float s = 0.0f, s2 = 0.0f;
  for (int r = 0; r < 256; ++r) {
    float x = H[(r0 + r) * F + f];           // threads across f: coalesced
    s += x; s2 += x * x;
  }
  atomicAdd(&sums[f], s);
  atomicAdd(&sums[F + f], s2);
}

// ---------------- BN apply -> next layer input buffer ---------------------------------
__global__ void bn_apply_kernel(const float* __restrict__ H, const float* __restrict__ sums,
                                const float* __restrict__ g, const float* __restrict__ be,
                                float* __restrict__ Out, int F, int Freal) {
  size_t idx = (size_t)blockIdx.x * blockDim.x + threadIdx.x;
  size_t total = (size_t)NM * F;
  if (idx >= total) return;
  int f = (int)(idx % F);
  if (f >= Freal) { Out[idx] = 0.0f; return; }
  const float invN = 1.0f / (float)NM;
  float m   = sums[f] * invN;
  float var = sums[F + f] * invN - m * m;
  Out[idx] = (H[idx] - m) * rsqrtf(var + 1e-5f) * g[f] + be[f];
}

// ---------------- global max-pool over nodes: pooled[b][f] ----------------------------
__global__ void maxpool_kernel(const float* __restrict__ H, float* __restrict__ pooled,
                               int Npad, int Freal) {
  __shared__ float red[256];
  int b = blockIdx.x / Freal, f = blockIdx.x % Freal;
  float mx = -1e30f;
  for (int v = threadIdx.x; v < NV; v += 256)
    mx = fmaxf(mx, H[((size_t)v * NB + b) * Npad + f]);
  red[threadIdx.x] = mx;
  __syncthreads();
  for (int s = 128; s > 0; s >>= 1) {
    if (threadIdx.x < s) red[threadIdx.x] = fmaxf(red[threadIdx.x], red[threadIdx.x + s]);
    __syncthreads();
  }
  if (threadIdx.x == 0) pooled[b * Freal + f] = red[0];
}

// ---------------- log_softmax over 10 classes -----------------------------------------
__global__ void logsoftmax_kernel(const float* __restrict__ pooled, float* __restrict__ out,
                                  int F) {
  int b = threadIdx.x;
  if (b >= NB) return;
  float mx = -1e30f;
  for (int f = 0; f < F; ++f) mx = fmaxf(mx, pooled[b * F + f]);
  float s = 0.0f;
  for (int f = 0; f < F; ++f) s += expf(pooled[b * F + f] - mx);
  float ls = logf(s);
  for (int f = 0; f < F; ++f) out[b * F + f] = pooled[b * F + f] - mx - ls;
}

extern "C" void kernel_launch(void* const* d_in, const int* in_sizes, int n_in,
                              void* d_out, int out_size, void* d_ws, size_t ws_size,
                              hipStream_t stream) {
  const float* x     = (const float*)d_in[0];
  const int*   lrow  = (const int*)d_in[1];
  const int*   lcol  = (const int*)d_in[2];
  const float* lval  = (const float*)d_in[3];
  const float* w_in  = (const float*)d_in[4];
  const float* b_in  = (const float*)d_in[5];
  const float* g0    = (const float*)d_in[6];
  const float* be0   = (const float*)d_in[7];
  const float* w_h0  = (const float*)d_in[8];
  const float* b_h0  = (const float*)d_in[9];
  const float* g1    = (const float*)d_in[10];
  const float* be1   = (const float*)d_in[11];
  const float* w_h1  = (const float*)d_in[12];
  const float* b_h1  = (const float*)d_in[13];
  const float* gout  = (const float*)d_in[14];
  const float* beout = (const float*)d_in[15];
  const float* w_out = (const float*)d_in[16];
  const float* b_out = (const float*)d_in[17];
  const int E = in_sizes[1];

  // workspace carve (3 rotating T buffers + acc, sized at max C=128 -> 64 MB each)
  char* p = (char*)d_ws;
  auto carve = [&](size_t bytes) -> char* {
    char* r = p; p += (bytes + 255) & ~(size_t)255; return r;
  };
  const size_t TBYTES = (size_t)NM * 128 * sizeof(float);
  float*  bufA   = (float*)carve(TBYTES);   // layer input / rotation
  float*  bufB   = (float*)carve(TBYTES);
  float*  bufC   = (float*)carve(TBYTES);
  float*  acc    = (float*)carve(TBYTES);   // GEMM accumulator [NM, Npad]
  __bf16* wpad   = (__bf16*)carve((size_t)KCHEB * 128 * 128 * sizeof(__bf16));
  float*  sums   = (float*)carve(2 * 128 * sizeof(float));
  float*  pooled = (float*)carve((size_t)NB * 16 * sizeof(float));

  // ---- one ChebConv layer: input in bufA as [NM, Cpad]; result -> acc (bias+relu'd)
  auto cheb_layer = [&](int Cpad, int Npad, int Nreal, int Cin_real,
                        const float* W, const float* bias) {
    size_t wtot = (size_t)KCHEB * Cpad * Npad;
    pad_weights_kernel<<<(unsigned)((wtot + 255) / 256), 256, 0, stream>>>(
        W, wpad, Cin_real, Nreal, Cpad, Npad);
    hipMemsetAsync(acc, 0, (size_t)NM * Npad * sizeof(float), stream);

    dim3 ggrid(NM / 128, Npad / 64);
    const size_t wk = (size_t)Cpad * Npad;
    const int ldc = NB * Cpad;
    const size_t tbytes = (size_t)NM * Cpad * sizeof(float);
    const size_t n4 = (size_t)NM * Cpad / 4;
    const long long sthreads = (long long)E * (ldc / 4);
    const unsigned sblocks = (unsigned)((sthreads + 255) / 256);

    float* t0 = bufA; float* t1 = bufB; float* t2 = bufC;

    // k = 0
    cheb_gemm_wmma<<<ggrid, 256, 0, stream>>>(t0, wpad, acc, Cpad, Npad, nullptr, Nreal);
    // k = 1 : T1 = L T0
    hipMemsetAsync(t1, 0, tbytes, stream);
    spmm_kernel<<<sblocks, 256, 0, stream>>>(lrow, lcol, lval, t0, t1, ldc, E);
    cheb_gemm_wmma<<<ggrid, 256, 0, stream>>>(t1, wpad + wk, acc, Cpad, Npad, nullptr, Nreal);
    // k = 2..5 : T2 = 2 L T1 - T0 ; fuse bias+ReLU into the final accumulation
    for (int k = 2; k < KCHEB; ++k) {
      hipMemsetAsync(t2, 0, tbytes, stream);
      spmm_kernel<<<sblocks, 256, 0, stream>>>(lrow, lcol, lval, t1, t2, ldc, E);
      cheb_combine_kernel<<<(unsigned)((n4 + 255) / 256), 256, 0, stream>>>(t2, t0, n4);
      const float* bptr = (k == KCHEB - 1) ? bias : nullptr;
      cheb_gemm_wmma<<<ggrid, 256, 0, stream>>>(t2, wpad + (size_t)k * wk, acc,
                                                Cpad, Npad, bptr, Nreal);
      float* tmp = t0; t0 = t1; t1 = t2; t2 = tmp;
    }
  };

  // ---- BN (train-mode stats) on acc -> bufA as next layer input
  auto bn_to_bufA = [&](int F, int Freal, const float* g, const float* be) {
    hipMemsetAsync(sums, 0, 2 * F * sizeof(float), stream);
    bn_stats_kernel<<<NM / 256, F, 0, stream>>>(acc, sums, F);
    size_t etot = (size_t)NM * F;
    bn_apply_kernel<<<(unsigned)((etot + 255) / 256), 256, 0, stream>>>(
        acc, sums, g, be, bufA, F, Freal);
  };

  // ---- pipeline --------------------------------------------------------------------
  {
    size_t etot = (size_t)NM * 32;
    transform_input_kernel<<<(unsigned)((etot + 255) / 256), 256, 0, stream>>>(x, bufA, 32);
  }
  cheb_layer(32, 64, 64, 3, w_in, b_in);           // 3 -> 64
  bn_to_bufA(64, 64, g0, be0);
  cheb_layer(64, 128, 128, 64, w_h0, b_h0);        // 64 -> 128
  bn_to_bufA(128, 128, g1, be1);
  cheb_layer(128, 128, 128, 128, w_h1, b_h1);      // 128 -> 128
  bn_to_bufA(128, 128, gout, beout);
  cheb_layer(128, 64, 10, 128, w_out, b_out);      // 128 -> 10 (padded N=64)

  maxpool_kernel<<<NB * 10, 256, 0, stream>>>(acc, pooled, 64, 10);
  logsoftmax_kernel<<<1, 32, 0, stream>>>(pooled, (float*)d_out, 10);
}
